// EpiLinear_88244398063978
// MI455X (gfx1250) — compile-verified, matrix operand order checked by reading, not run on hardware
//
#include <hip/hip_runtime.h>
#include <hip/hip_bf16.h>

// ---------- CDNA5 WMMA types ----------
typedef __attribute__((ext_vector_type(16))) __bf16 v16bf;
typedef __attribute__((ext_vector_type(2)))  __bf16 v2bf;
typedef __attribute__((ext_vector_type(8)))  float  v8f;

struct __align__(16) U4 { unsigned int x, y, z, w; };

union Frag {
    v16bf v;
    unsigned int w[8];
    U4 q[2];
};

// Native fptrunc (round-to-nearest-even) — hardware bf16 conversion.
__device__ __forceinline__ unsigned short f32_to_bf16(float f) {
    union { __bf16 b; unsigned short u; } t;
    t.b = (__bf16)f;
    return t.u;
}

// One v_cvt_pk_bf16_f32 when the builtin exists; no half re-packing.
__device__ __forceinline__ unsigned int pack_bf16x2(float lo, float hi) {
#if __has_builtin(__builtin_amdgcn_cvt_pk_bf16_f32)
    union { v2bf v; unsigned int u; } t;
    t.v = __builtin_amdgcn_cvt_pk_bf16_f32(lo, hi);
    return t.u;
#else
    union { __bf16 b[2]; unsigned int u; } t;
    t.b[0] = (__bf16)lo;
    t.b[1] = (__bf16)hi;
    return t.u;
#endif
}

// Constants from the reference
#define BATCH   2048
#define NZ      8
#define NDIM    32
#define SDIM    1024
#define HDIM    1024
#define EPIH    512
#define XFK     2048      // shared part of epinet input (x ++ feature)

// ============================================================
// Kernel D: one-time bf16 weight packing into fragment-friendly
// (K-contiguous) layouts. Wep1P[n][k] (transposed), W1zP[j][k],
// W2P[j][c] (Wep2 row-major already matches).
// ============================================================
__global__ __launch_bounds__(256) void pack_weights_kernel(
    const float* __restrict__ Wep1, const float* __restrict__ Wep2,
    unsigned short* __restrict__ Wep1P, unsigned short* __restrict__ W1zP,
    unsigned short* __restrict__ W2P)
{
    const int stride = gridDim.x * 256;
    const int t0 = blockIdx.x * 256 + threadIdx.x;
    for (int i = t0; i < EPIH * XFK; i += stride) {
        int n = i >> 11, k = i & 2047;                 // Wep1P[n*2048+k]
        Wep1P[i] = f32_to_bf16(Wep1[k * EPIH + n]);
    }
    for (int i = t0; i < EPIH * 32; i += stride) {
        int j = i >> 5, k = i & 31;                    // W1zP[j*32+k]
        W1zP[i] = f32_to_bf16(Wep1[(XFK + k) * EPIH + j]);
        W2P[i]  = f32_to_bf16(Wep2[i]);                // [j][c] identical layout
    }
}

// ============================================================
// Kernel A: H0[2048,512] = concat(x,feature) @ Wep1[0:2048,:]
// bf16 WMMA, block tile 128(M) x 64(N), K-step 32.
// K loop split into the x-half and feature-half so the source
// select is loop-invariant; all addresses are running pointers
// with constant-immediate offsets for the unrolled bodies.
// LDS rows are 80B stride: 16B-aligned b128 fragment loads,
// 20-bank stride => conflict-free across the 16 fragment lanes.
// ============================================================
__global__ __launch_bounds__(256) void epi_gemm_kernel(
    const float* __restrict__ x, const float* __restrict__ feat,
    const unsigned short* __restrict__ Wep1P, float* __restrict__ H0)
{
    __shared__ __align__(16) unsigned short As[128][40];
    __shared__ __align__(16) unsigned short Bs[64][40];

    const int tid  = threadIdx.x;
    const int wave = tid >> 5;        // 0..7 -> M sub-tile
    const int lane = tid & 31;
    const int r    = lane & 15;
    const int hf   = lane >> 4;
    const int row_blk = blockIdx.y * 128;
    const int col_blk = blockIdx.x * 64;

    // ---- staging coordinates (constant across the whole K loop) ----
    const int arow = tid >> 4;                 // 0..15 (A rows i*16+arow)
    const int acol = (tid & 15) << 1;          // even k, 0..30
    unsigned int* as_st = reinterpret_cast<unsigned int*>(&As[arow][acol]);
    U4* bs_st = reinterpret_cast<U4*>(&Bs[tid >> 2][(tid & 3) * 8]);
    const unsigned short* pb =
        Wep1P + (col_blk + (tid >> 2)) * XFK + (tid & 3) * 8;

    // ---- fragment LDS read pointers (constant) ----
    const U4* a_ld = reinterpret_cast<const U4*>(&As[wave * 16 + r][hf * 8]);
    const U4* b_ld = reinterpret_cast<const U4*>(&Bs[r][hf * 8]);

    const v8f vzero = {0.f,0.f,0.f,0.f,0.f,0.f,0.f,0.f};
    v8f acc[4];
#pragma unroll
    for (int t = 0; t < 4; ++t) acc[t] = vzero;

#pragma unroll 1
    for (int half = 0; half < 2; ++half) {
        const float* base = half ? feat : x;
        const float* pa = base + (row_blk + arow) * SDIM + acol;
#pragma unroll 1
        for (int k0 = 0; k0 < SDIM; k0 += 32) {
            // stage A tile 128x32: 8 float2 loads at constant row offsets
#pragma unroll
            for (int i = 0; i < 8; ++i) {
                float2 f = *reinterpret_cast<const float2*>(pa + i * 16 * SDIM);
                as_st[i * 320] = pack_bf16x2(f.x, f.y);   // 320 uints = 16 rows
            }
            // stage B tile 64(n) x 32(k): one b128 copy per thread
            *bs_st = *reinterpret_cast<const U4*>(pb);
            // prefetch next K tile (global_prefetch_b8)
            if (k0 + 32 < SDIM) __builtin_prefetch(pa + 32, 0, 1);
            __syncthreads();

            Frag a;                    // two ds_load_b128
            a.q[0] = a_ld[0];
            a.q[1] = a_ld[2];
#pragma unroll
            for (int t = 0; t < 4; ++t) {
                Frag b;
                b.q[0] = b_ld[t * 80];       // t*16 rows * 40 ushort = 80 U4
                b.q[1] = b_ld[t * 80 + 2];
                acc[t] = __builtin_amdgcn_wmma_f32_16x16x32_bf16(
                    false, a.v, false, b.v, (short)0, acc[t], false, false);
            }
            __syncthreads();
            pa += 32;
            pb += 32;
        }
    }

    // D layout: lane holds (m = v + 8*hf, n = r)
    float* ho = H0 + (row_blk + wave * 16 + 8 * hf) * EPIH + col_blk + r;
#pragma unroll
    for (int t = 0; t < 4; ++t)
#pragma unroll
        for (int v = 0; v < 8; ++v)
            ho[v * EPIH + t * 16] = acc[t][v];
}

// ============================================================
// Kernel B: prior ensemble (32 tiny MLPs 1024->5->5->1) -> p[2048,32]
// One block per batch row; x row staged in LDS. Pure VALU (0.67 GFLOP).
// ============================================================
__global__ __launch_bounds__(256) void prior_kernel(
    const float* __restrict__ x,
    const float* __restrict__ Wp1, const float* __restrict__ bp1,
    const float* __restrict__ Wp2, const float* __restrict__ bp2,
    const float* __restrict__ Wp3, const float* __restrict__ bp3,
    float* __restrict__ p)
{
    __shared__ float xs[1024];
    __shared__ float h1s[160];
    __shared__ float h2s[160];

    const int b   = blockIdx.x;
    const int tid = threadIdx.x;

    for (int i = tid; i < SDIM; i += 256) xs[i] = x[b * SDIM + i];
    __syncthreads();

    if (tid < 160) {                    // layer 1: [1024]->[32,5]
        int e = tid / 5, h = tid % 5;
        float s = bp1[e * 5 + h];
        const float* w = Wp1 + e * SDIM * 5 + h;
        for (int k = 0; k < SDIM; ++k) s += xs[k] * w[k * 5];
        h1s[tid] = s > 0.f ? s : 0.f;
    }
    __syncthreads();

    if (tid < 160) {                    // layer 2: [32,5]->[32,5]
        int e = tid / 5, g = tid % 5;
        float s = bp2[e * 5 + g];
#pragma unroll
        for (int h = 0; h < 5; ++h) s += h1s[e * 5 + h] * Wp2[e * 25 + h * 5 + g];
        h2s[tid] = s > 0.f ? s : 0.f;
    }
    __syncthreads();

    if (tid < 32) {                     // layer 3: [32,5]->[32,1]
        float s = bp3[tid];
#pragma unroll
        for (int g = 0; g < 5; ++g) s += h2s[tid * 5 + g] * Wp3[tid * 5 + g];
        p[b * 32 + tid] = s;
    }
}

// ============================================================
// Kernel C: fused tail. Per wave, 16 rows (row = b*8 + n).
//   pre = Z@Wep1[2048:] + H0[b] + bep1 ; wz = Z@Wep2^T
//   out[row] = sum_j relu(pre)*wz  +  z[row] . (bep2 + p[b])
// Both GEMMs share the same A fragment (Z, K=32 -> 1 WMMA each);
// B fragments are two global_load_b128 each from packed bf16
// weights, walked with running pointers.
// ============================================================
__global__ __launch_bounds__(256) void epi_tail_kernel(
    const float* __restrict__ z,
    const unsigned short* __restrict__ W1zP,
    const unsigned short* __restrict__ W2P,
    const float* __restrict__ bep1, const float* __restrict__ bep2,
    const float* __restrict__ H0, const float* __restrict__ p,
    float* __restrict__ out)
{
    __shared__ float rowsum[8][16];

    const int tid  = threadIdx.x;
    const int wave = tid >> 5;
    const int lane = tid & 31;
    const int r    = lane & 15;
    const int hf   = lane >> 4;
    const int row_base = (blockIdx.x * 8 + wave) * 16;   // 16 rows per wave
    const int b_lane   = (row_base >> 3) + hf;           // b for this lane's D rows

    // A fragment: 16 rows of Z (K = 32 exactly), packed from wide reads
    Frag a;
    {
        const float* zr = z + (row_base + r) * NDIM;
#pragma unroll
        for (int h = 0; h < 2; ++h) {
            const float* src = zr + h * 16 + hf * 8;
#pragma unroll
            for (int i = 0; i < 4; ++i)
                a.w[h * 4 + i] = pack_bf16x2(src[2 * i], src[2 * i + 1]);
        }
    }

    // running pointers for the j loop
    const unsigned short* p1 = W1zP + r * 32 + hf * 8;
    const unsigned short* p2 = W2P  + r * 32 + hf * 8;
    const float* ph  = H0 + b_lane * EPIH + r;
    const float* pbias = bep1 + r;

    const v8f vzero = {0.f,0.f,0.f,0.f,0.f,0.f,0.f,0.f};
    float acc[8];
#pragma unroll
    for (int v = 0; v < 8; ++v) acc[v] = 0.f;

#pragma unroll 1
    for (int j0 = 0; j0 < EPIH; j0 += 16) {
        Frag b1, b2;
        b1.q[0] = *reinterpret_cast<const U4*>(p1);
        b1.q[1] = *reinterpret_cast<const U4*>(p1 + 16);
        b2.q[0] = *reinterpret_cast<const U4*>(p2);
        b2.q[1] = *reinterpret_cast<const U4*>(p2 + 16);

        v8f c1 = __builtin_amdgcn_wmma_f32_16x16x32_bf16(
            false, a.v, false, b1.v, (short)0, vzero, false, false);
        v8f c2 = __builtin_amdgcn_wmma_f32_16x16x32_bf16(
            false, a.v, false, b2.v, (short)0, vzero, false, false);

        // shared-part preactivation + bias: depends only on (b_lane, col)
        float h0v = *ph + *pbias;
#pragma unroll
        for (int v = 0; v < 8; ++v) {
            float pre = c1[v] + h0v;
            pre = pre > 0.f ? pre : 0.f;
            acc[v] += pre * c2[v];
        }
        p1 += 16 * 32;
        p2 += 16 * 32;
        ph += 16;
        pbias += 16;
    }

    // reduce over the 16 n-lanes within each half (rows stay separated)
#pragma unroll
    for (int off = 1; off < 16; off <<= 1)
#pragma unroll
        for (int v = 0; v < 8; ++v)
            acc[v] += __shfl_xor(acc[v], off, 32);

    if (r == 0) {
#pragma unroll
        for (int v = 0; v < 8; ++v) rowsum[wave][hf * 8 + v] = acc[v];
    }
    __syncthreads();

    if (lane < 16) {
        int row = row_base + lane;
        int b   = row >> 3;
        float t = 0.f;
#pragma unroll
        for (int c = 0; c < 32; ++c)
            t += z[row * NDIM + c] * (bep2[c] + p[b * 32 + c]);
        out[row] = rowsum[wave][lane] + t;
    }
}

// ============================================================
extern "C" void kernel_launch(void* const* d_in, const int* in_sizes, int n_in,
                              void* d_out, int out_size, void* d_ws, size_t ws_size,
                              hipStream_t stream) {
    const float* x       = (const float*)d_in[0];
    const float* feature = (const float*)d_in[1];
    const float* z       = (const float*)d_in[2];
    const float* Wep1    = (const float*)d_in[3];
    const float* bep1    = (const float*)d_in[4];
    const float* Wep2    = (const float*)d_in[5];
    const float* bep2    = (const float*)d_in[6];
    const float* Wp1     = (const float*)d_in[7];
    const float* bp1     = (const float*)d_in[8];
    const float* Wp2     = (const float*)d_in[9];
    const float* bp2     = (const float*)d_in[10];
    const float* Wp3     = (const float*)d_in[11];
    const float* bp3     = (const float*)d_in[12];

    float* H0 = (float*)d_ws;                          // [2048, 512] f32
    float* pw = H0 + BATCH * EPIH;                     // [2048, 32]  f32
    unsigned short* Wep1P = (unsigned short*)(pw + BATCH * 32); // [512][2048] bf16
    unsigned short* W1zP  = Wep1P + EPIH * XFK;        // [512][32] bf16
    unsigned short* W2P   = W1zP + EPIH * 32;          // [512][32] bf16
    float* out = (float*)d_out;                        // [2048, 8]

    pack_weights_kernel<<<512, 256, 0, stream>>>(Wep1, Wep2, Wep1P, W1zP, W2P);
    epi_gemm_kernel<<<dim3(EPIH / 64, BATCH / 128), 256, 0, stream>>>(
        x, feature, Wep1P, H0);
    prior_kernel<<<BATCH, 256, 0, stream>>>(
        x, Wp1, bp1, Wp2, bp2, Wp3, bp3, pw);
    epi_tail_kernel<<<(BATCH * NZ) / 128, 256, 0, stream>>>(
        z, W1zP, W2P, bep1, bep2, H0, pw, out);
}
